// LocalAttention_2439541424208
// MI455X (gfx1250) — compile-verified, matrix operand order checked by reading, not run on hardware
//
#include <hip/hip_runtime.h>
#include <hip/hip_bf16.h>

typedef __attribute__((ext_vector_type(16))) __bf16 v16bf;
typedef __attribute__((ext_vector_type(8)))  float  v8f;

#define T_LEN 4096
#define E 64
#define WIN 128
#define KW 256
#define QK_STRIDE 72    // 64 + 8 pad (bf16 elems), row = 144B (16B aligned)
#define VT_STRIDE 264   // 256 + 8 pad, row = 528B (16B aligned)
#define P_STRIDE 40     // 32 + 8 pad, row = 80B (16B aligned)

union BF4 { __bf16 h[4]; uint2 u2; };

__global__ __launch_bounds__(256) void local_attn_rope_kernel(
    const float* __restrict__ qg, const float* __restrict__ kg,
    const float* __restrict__ vg, float* __restrict__ outg)
{
    __shared__ __bf16 Qb[WIN * QK_STRIDE];        // 18 KB
    __shared__ __bf16 Kb[KW  * QK_STRIDE];        // 36 KB
    __shared__ __bf16 Vt[E   * VT_STRIDE];        // 33 KB
    __shared__ __bf16 Ps[8 * 16 * P_STRIDE];      // 10 KB per-wave P scratch

    const int tid = threadIdx.x;
    const int w   = blockIdx.x;        // window 0..31
    const int bh  = blockIdx.y;        // 0..63
    const long base = (long)bh * T_LEN * E;
    const float LN1E4_32 = 0.28782313662425572f;  // ln(10000)/32

    // ---- Stage Q with RoPE -> bf16 LDS: 128 rows x 8 groups of 4 pairs ----
#pragma unroll
    for (int it = 0; it < 4; ++it) {
        int idx = tid + it * 256;
        int row = idx >> 3, g = idx & 7, p0 = g * 4;
        int t = w * WIN + row;
        const float* src = qg + base + (long)t * E;
        float4 x1 = *(const float4*)(src + p0);
        float4 x2 = *(const float4*)(src + p0 + 32);
        BF4 a, b;
#pragma unroll
        for (int e = 0; e < 4; ++e) {
            float xa = ((const float*)&x1)[e], xb = ((const float*)&x2)[e];
            float f = (float)t * __expf(-(float)(p0 + e) * LN1E4_32);
            float c = __cosf(f), s = __sinf(f);
            a.h[e] = (__bf16)(xa * c - xb * s);
            b.h[e] = (__bf16)(xb * c + xa * s);
        }
        *(uint2*)(&Qb[row * QK_STRIDE + p0])      = a.u2;
        *(uint2*)(&Qb[row * QK_STRIDE + p0 + 32]) = b.u2;
    }
    // ---- Stage K with RoPE (256 rows incl. look-back; OOB -> 0) ----
#pragma unroll
    for (int it = 0; it < 8; ++it) {
        int idx = tid + it * 256;
        int row = idx >> 3, g = idx & 7, p0 = g * 4;
        int t = w * WIN - WIN + row;
        float4 x1 = make_float4(0.f, 0.f, 0.f, 0.f);
        float4 x2 = make_float4(0.f, 0.f, 0.f, 0.f);
        if (t >= 0) {
            const float* src = kg + base + (long)t * E;
            x1 = *(const float4*)(src + p0);
            x2 = *(const float4*)(src + p0 + 32);
        }
        BF4 a, b;
#pragma unroll
        for (int e = 0; e < 4; ++e) {
            float xa = ((const float*)&x1)[e], xb = ((const float*)&x2)[e];
            float f = (float)t * __expf(-(float)(p0 + e) * LN1E4_32);
            float c = __cosf(f), s = __sinf(f);
            a.h[e] = (__bf16)(xa * c - xb * s);
            b.h[e] = (__bf16)(xb * c + xa * s);
        }
        *(uint2*)(&Kb[row * QK_STRIDE + p0])      = a.u2;
        *(uint2*)(&Kb[row * QK_STRIDE + p0 + 32]) = b.u2;
    }
    // ---- Stage V transposed: Vt[e][key]; float4 global loads ----
#pragma unroll
    for (int it = 0; it < 16; ++it) {
        int idx = tid + it * 256;
        int row = idx >> 4, cg = idx & 15, c0 = cg * 4;
        int t = w * WIN - WIN + row;
        float4 x = make_float4(0.f, 0.f, 0.f, 0.f);
        if (t >= 0) x = *(const float4*)(vg + base + (long)t * E + c0);
#pragma unroll
        for (int e = 0; e < 4; ++e)
            Vt[(c0 + e) * VT_STRIDE + row] = (__bf16)(((const float*)&x)[e]);
    }
    __syncthreads();

    const int lane = tid & 31;
    const int wv   = tid >> 5;      // wave id 0..7
    const int hi   = lane >> 4;     // half-wave
    const int ln   = lane & 15;
    const int i0   = wv * 16;       // query-row base for this wave

    // Q A-fragments (16x32 bf16 each), two K-chunks covering e=64
    v16bf qa[2];
#pragma unroll
    for (int c = 0; c < 2; ++c) {
        const __bf16* qrow = &Qb[(i0 + ln) * QK_STRIDE];
        int koff = c * 32 + hi * 8;
        ((uint4*)&qa[c])[0] = *(const uint4*)(qrow + koff);
        ((uint4*)&qa[c])[1] = *(const uint4*)(qrow + koff + 16);
    }

    float mrow[8], lrow[8];
    v8f oacc[4];
#pragma unroll
    for (int r = 0; r < 8; ++r) { mrow[r] = -1e30f; lrow[r] = 0.f; }
#pragma unroll
    for (int n = 0; n < 4; ++n)
#pragma unroll
        for (int r = 0; r < 8; ++r) oacc[n][r] = 0.f;

    __bf16* pscr = &Ps[wv * 16 * P_STRIDE];

    // ---- Flash loop over 8 chunks of 32 keys ----
    for (int jp = 0; jp < 8; ++jp) {
        const int j0 = jp * 32;
        v8f accA, accB;
#pragma unroll
        for (int r = 0; r < 8; ++r) { accA[r] = 0.f; accB[r] = 0.f; }
#pragma unroll
        for (int c = 0; c < 2; ++c) {
            int koff = c * 32 + hi * 8;
            v16bf bA, bB;
            const __bf16* krA = &Kb[(j0 + ln) * QK_STRIDE];
            const __bf16* krB = &Kb[(j0 + 16 + ln) * QK_STRIDE];
            ((uint4*)&bA)[0] = *(const uint4*)(krA + koff);
            ((uint4*)&bA)[1] = *(const uint4*)(krA + koff + 16);
            ((uint4*)&bB)[0] = *(const uint4*)(krB + koff);
            ((uint4*)&bB)[1] = *(const uint4*)(krB + koff + 16);
            accA = __builtin_amdgcn_wmma_f32_16x16x32_bf16(false, qa[c], false, bA, (short)0, accA, false, false);
            accB = __builtin_amdgcn_wmma_f32_16x16x32_bf16(false, qa[c], false, bB, (short)0, accB, false, false);
        }
        // scale + causal/pad mask + per-row tile max
        float tmax[8];
#pragma unroll
        for (int r = 0; r < 8; ++r) {
            int irow = i0 + r + hi * 8;
            int jA = j0 + ln, jB = j0 + 16 + ln;
            float sA = accA[r] * 0.125f;
            float sB = accB[r] * 0.125f;
            if ((jA > irow + 128) || (w == 0 && jA < 128)) sA = -1e30f;
            if ((jB > irow + 128) || (w == 0 && jB < 128)) sB = -1e30f;
            accA[r] = sA; accB[r] = sB;
            float t = fmaxf(sA, sB);
            t = fmaxf(t, __shfl_xor(t, 1));
            t = fmaxf(t, __shfl_xor(t, 2));
            t = fmaxf(t, __shfl_xor(t, 4));
            t = fmaxf(t, __shfl_xor(t, 8));
            tmax[r] = t;
        }
        // online softmax update + write bf16 P tile to per-wave scratch
#pragma unroll
        for (int r = 0; r < 8; ++r) {
            float mnew  = fmaxf(fmaxf(mrow[r], tmax[r]), -1e29f);
            float alpha = __expf(mrow[r] - mnew);
            float pA = __expf(accA[r] - mnew);
            float pB = __expf(accB[r] - mnew);
            float ps = pA + pB;
            ps += __shfl_xor(ps, 1);
            ps += __shfl_xor(ps, 2);
            ps += __shfl_xor(ps, 4);
            ps += __shfl_xor(ps, 8);
            lrow[r] = lrow[r] * alpha + ps;
            mrow[r] = mnew;
#pragma unroll
            for (int n = 0; n < 4; ++n) oacc[n][r] *= alpha;
            int prow = r + hi * 8;
            pscr[prow * P_STRIDE + ln]      = (__bf16)pA;
            pscr[prow * P_STRIDE + 16 + ln] = (__bf16)pB;
        }
        asm volatile("" ::: "memory");   // keep DS store->load order (same wave, DS in-order)

        // P A-fragment (16x32) from scratch, then PV WMMAs
        v16bf pf;
        {
            const __bf16* pr = pscr + ln * P_STRIDE + hi * 8;
            ((uint4*)&pf)[0] = *(const uint4*)(pr);
            ((uint4*)&pf)[1] = *(const uint4*)(pr + 16);
        }
#pragma unroll
        for (int n = 0; n < 4; ++n) {
            v16bf vf;
            const __bf16* vr = &Vt[(n * 16 + ln) * VT_STRIDE + j0];
            ((uint4*)&vf)[0] = *(const uint4*)(vr + hi * 8);
            ((uint4*)&vf)[1] = *(const uint4*)(vr + hi * 8 + 16);
            oacc[n] = __builtin_amdgcn_wmma_f32_16x16x32_bf16(false, pf, false, vf, (short)0, oacc[n], false, false);
        }
    }

    // ---- normalize and write out (f32) ----
#pragma unroll
    for (int n = 0; n < 4; ++n)
#pragma unroll
        for (int r = 0; r < 8; ++r) {
            int irow = i0 + r + hi * 8;
            int col  = n * 16 + ln;
            outg[base + (long)(w * WIN + irow) * E + col] = oacc[n][r] / lrow[r];
        }
}

extern "C" void kernel_launch(void* const* d_in, const int* in_sizes, int n_in,
                              void* d_out, int out_size, void* d_ws, size_t ws_size,
                              hipStream_t stream) {
    const float* q = (const float*)d_in[0];
    const float* k = (const float*)d_in[1];
    const float* v = (const float*)d_in[2];
    float* out = (float*)d_out;
    dim3 grid(T_LEN / WIN, 64);   // (32 windows, B*H)
    local_attn_rope_kernel<<<grid, 256, 0, stream>>>(q, k, v, out);
}
